// SmallWorldNodeEmbedder_6330781794648
// MI455X (gfx1250) — compile-verified, matrix operand order checked by reading, not run on hardware
//
#include <hip/hip_runtime.h>

#define E_DIM 256
#define IN_DIM 257
#define B_DIM 128
#define N_DIM 1027

typedef __bf16 v16bf __attribute__((ext_vector_type(16)));
typedef float v8f    __attribute__((ext_vector_type(8)));
typedef unsigned v8u __attribute__((ext_vector_type(8)));

// ---------------------------------------------------------------------------
// Workspace layout (bytes):
//   [0,          524288)  : W_hi plane, WMMA-fragment order [s][kc][ct][L][v]
//   [524288,    1048576)  : W_lo plane, same order
//   [1048576,   1052672)  : W tail column (k=256), fp32 [seg][n]
//   [1052672,   1056768)  : fused bias, fp32 [seg][n]
// Fragment order: u32 index g = (((s*8+kc)*16+ct)*32+L)*8+v packs the bf16
// pair (k0, k0+1) for column n=ct*16+(L&15), k0=kc*32+2*j,
// j = (v<4) ? (L>>4)*4+v : 8+(L>>4)*4+(v-4)   (ISA 7.12.2 B layout dual of A)
// ---------------------------------------------------------------------------

__device__ __forceinline__ float eff_w(const float* __restrict__ W,
                                       const float* __restrict__ Wreg,
                                       const float* __restrict__ Wrand,
                                       float alpha, int idx) {
    return (1.0f - alpha) * W[idx] + alpha * (Wreg[idx] + 0.1f * Wrand[idx]);
}

__global__ __launch_bounds__(256) void prep_frag_kernel(
    const float* __restrict__ Wsrc, const float* __restrict__ Wcand,
    const float* __restrict__ Wdst, const float* __restrict__ Wdep,
    const float* __restrict__ Wreg, const float* __restrict__ Wrand,
    const float* __restrict__ iw,
    unsigned* __restrict__ whP, unsigned* __restrict__ wlP)
{
    const int g  = blockIdx.x * 256 + threadIdx.x;   // 0..131071
    const int v  = g & 7;
    const int L  = (g >> 3) & 31;
    const int ct = (g >> 8) & 15;
    const int kc = (g >> 12) & 7;
    const int s  = g >> 15;
    const int half = L >> 4, nl = L & 15;
    const int j  = (v < 4) ? (half * 4 + v) : (8 + half * 4 + (v - 4));
    const int n  = ct * 16 + nl;
    const int k0 = kc * 32 + 2 * j;

    const float alpha = 1.0f / (1.0f + __expf(-iw[0]));
    const float* W = (s == 0) ? Wsrc : (s == 1) ? Wcand : (s == 2) ? Wdst : Wdep;

    const float w0 = eff_w(W, Wreg, Wrand, alpha, n * IN_DIM + k0);
    const float w1 = eff_w(W, Wreg, Wrand, alpha, n * IN_DIM + k0 + 1);
    const __bf16 h0 = (__bf16)w0, h1 = (__bf16)w1;
    const __bf16 l0 = (__bf16)(w0 - (float)h0);
    const __bf16 l1 = (__bf16)(w1 - (float)h1);
    whP[g] = ((unsigned)__builtin_bit_cast(unsigned short, h1) << 16)
           |  (unsigned)__builtin_bit_cast(unsigned short, h0);
    wlP[g] = ((unsigned)__builtin_bit_cast(unsigned short, l1) << 16)
           |  (unsigned)__builtin_bit_cast(unsigned short, l0);
}

__global__ __launch_bounds__(256) void prep_tail_kernel(
    const float* __restrict__ Wsrc, const float* __restrict__ bsrc,
    const float* __restrict__ Wcand,const float* __restrict__ bcand,
    const float* __restrict__ Wdst, const float* __restrict__ bdst,
    const float* __restrict__ Wdep, const float* __restrict__ bdep,
    const float* __restrict__ Wreg, const float* __restrict__ breg,
    const float* __restrict__ Wrand,const float* __restrict__ iw,
    float* __restrict__ wtail, float* __restrict__ bias)
{
    const int s = blockIdx.x;        // 0..3
    const int n = threadIdx.x;       // 0..255
    const float alpha = 1.0f / (1.0f + __expf(-iw[0]));
    const float* W = (s == 0) ? Wsrc : (s == 1) ? Wcand : (s == 2) ? Wdst : Wdep;
    const float* b = (s == 0) ? bsrc : (s == 1) ? bcand : (s == 2) ? bdst : bdep;
    wtail[s * 256 + n] = eff_w(W, Wreg, Wrand, alpha, n * IN_DIM + 256);
    bias[s * 256 + n]  = (1.0f - alpha) * b[n] + alpha * breg[n];
}

// ---------------------------------------------------------------------------
// GEMM: out[seg] = X[seg] (M x 257) * W_eff[seg]^T (257 x 256) + bias
// Block tile 64 x 256, 8 wave32 waves. Async global->LDS staging:
//   X: b32 per-lane copy, double-buffered (HBM stream overlaps WMMA)
//   W: b128 linear copy of pre-permuted fragments (L2-resident)
// B fragments load as 2x ds_load_b128 into contiguous VGPR quads.
// ---------------------------------------------------------------------------

__global__ __launch_bounds__(256) void gemm_kernel(
    const float* __restrict__ sol,
    const unsigned* __restrict__ whP,
    const unsigned* __restrict__ wlP,
    const float* __restrict__ wtail,
    const float* __restrict__ bias,
    float* __restrict__ out)
{
    __shared__ float xt[2][64 * 33];                  // X chunk dbl buffer
    __shared__ float xtail[64];                       // X column k=256
    __shared__ __align__(16) unsigned whB[16 * 256];  // W_hi frags, 16KB
    __shared__ __align__(16) unsigned wlB[16 * 256];  // W_lo frags, 16KB

    // ---- segment / row mapping ------------------------------------------
    const int bid = blockIdx.x;                       // 2+2048+2+2 blocks
    int s, segRow;
    if (bid < 2)         { s = 0; segRow = bid << 6; }
    else if (bid < 2050) { s = 1; segRow = (bid - 2) << 6; }
    else if (bid < 2052) { s = 2; segRow = (bid - 2050) << 6; }
    else                 { s = 3; segRow = (bid - 2052) << 6; }

    const int tokStart = (s == 0) ? 0 : (s == 1) ? 1 : (s == 2) ? (N_DIM - 2) : (N_DIM - 1);
    long growBase, rowStride;
    if (s == 1) {
        const int b0 = segRow >> 10, tok0 = segRow & 1023;
        growBase  = (long)b0 * N_DIM + tokStart + tok0;
        rowStride = IN_DIM;
    } else {
        growBase  = (long)segRow * N_DIM + tokStart;
        rowStride = (long)N_DIM * IN_DIM;
    }
    const float* xbase = sol + growBase * IN_DIM;

    const int t    = threadIdx.x;
    const int lane = t & 31;
    const int wid  = t >> 5;
    const int wm   = wid >> 1;       // 0..3 : row tile
    const int wn   = wid & 1;        // 0..1 : col half
    const int half = lane >> 4;
    const int nl   = lane & 15;
    const int ml   = lane & 15;

    if (t < 64) xtail[t] = xbase[(long)t * rowStride + 256];

    v8f c[8];
#pragma unroll
    for (int i = 0; i < 8; ++i) c[i] = (v8f){0, 0, 0, 0, 0, 0, 0, 0};

    const unsigned xtLds0 = (unsigned)(size_t)&xt[0][0];
    const unsigned xtLds1 = (unsigned)(size_t)&xt[1][0];
    const unsigned whLds  = (unsigned)(size_t)&whB[0];
    const unsigned wlLds  = (unsigned)(size_t)&wlB[0];

    // 8 async b32 per wave: one X chunk (64 rows x 32 k, rows not 16B-aligned)
    auto stage_x = [&](int kcn, int buf) {
        const unsigned base = buf ? xtLds1 : xtLds0;
#pragma unroll
        for (int ii = 0; ii < 8; ++ii) {
            const int i = t + ii * 256;
            const int r = i >> 5, k = i & 31;
            const unsigned lds = base + (unsigned)((r * 33 + k) * 4);
            const float* g = xbase + (long)r * rowStride + kcn * 32 + k;
            asm volatile("global_load_async_to_lds_b32 %0, %1, off"
                         :: "v"(lds), "v"(g) : "memory");
        }
    };
    // 8 async b128 per wave: linear copy of pre-permuted W fragments (16KB x2)
    auto stage_w = [&](int kcn) {
        const unsigned* gph = whP + ((unsigned)(s * 8 + kcn) << 12);
        const unsigned* gpl = wlP + ((unsigned)(s * 8 + kcn) << 12);
#pragma unroll
        for (int ii = 0; ii < 4; ++ii) {
            const int i = t + ii * 256;          // b128 slot 0..1023
            const unsigned dsth = whLds + (unsigned)(i * 16);
            const unsigned dstl = wlLds + (unsigned)(i * 16);
            asm volatile("global_load_async_to_lds_b128 %0, %1, off"
                         :: "v"(dsth), "v"(gph + i * 4) : "memory");
            asm volatile("global_load_async_to_lds_b128 %0, %1, off"
                         :: "v"(dstl), "v"(gpl + i * 4) : "memory");
        }
    };

    stage_x(0, 0);   // prologue: X chunk 0 in flight

    for (int kc = 0; kc < 8; ++kc) {
        const int cur = kc & 1;
        stage_w(kc);                          // W[kc] (8 ops)
        if (kc < 7) {
            stage_x(kc + 1, cur ^ 1);         // X[kc+1] prefetch (8 ops)
            asm volatile("s_wait_asynccnt 0x8" ::: "memory"); // X[kc],W[kc] done
        } else {
            asm volatile("s_wait_asynccnt 0x0" ::: "memory");
        }
        __syncthreads();

        // ---- A fragments: fp32 -> bf16 hi + lo split --------------------
        const float* xrow = &xt[cur][(wm * 16 + ml) * 33];
        v8u uah, ual;
#pragma unroll
        for (int v = 0; v < 8; ++v) {
            const int k0 = (v < 4) ? (half * 8 + 2 * v)
                                   : (16 + half * 8 + 2 * (v - 4));
            const float x0 = xrow[k0], x1 = xrow[k0 + 1];
            const __bf16 h0 = (__bf16)x0, h1 = (__bf16)x1;
            const __bf16 l0 = (__bf16)(x0 - (float)h0);
            const __bf16 l1 = (__bf16)(x1 - (float)h1);
            uah[v] = ((unsigned)__builtin_bit_cast(unsigned short, h1) << 16)
                   |  (unsigned)__builtin_bit_cast(unsigned short, h0);
            ual[v] = ((unsigned)__builtin_bit_cast(unsigned short, l1) << 16)
                   |  (unsigned)__builtin_bit_cast(unsigned short, l0);
        }
        const v16bf ah = __builtin_bit_cast(v16bf, uah);
        const v16bf al = __builtin_bit_cast(v16bf, ual);

#pragma unroll
        for (int tt = 0; tt < 8; ++tt) {
            const int ct = wn * 8 + tt;
            // lane's fragment slab: 8 contiguous u32 (32B) -> 2x ds_load_b128
            const v8u ubh = *(const v8u*)(whB + (ct * 32 + lane) * 8);
            const v8u ubl = *(const v8u*)(wlB + (ct * 32 + lane) * 8);
            const v16bf bh = __builtin_bit_cast(v16bf, ubh);
            const v16bf bl = __builtin_bit_cast(v16bf, ubl);
            // x*W ~= xh*Wh + xh*Wl + xl*Wh (fp32 accumulate)
            c[tt] = __builtin_amdgcn_wmma_f32_16x16x32_bf16(false, ah, false, bh, (short)0, c[tt], false, false);
            c[tt] = __builtin_amdgcn_wmma_f32_16x16x32_bf16(false, ah, false, bl, (short)0, c[tt], false, false);
            c[tt] = __builtin_amdgcn_wmma_f32_16x16x32_bf16(false, al, false, bh, (short)0, c[tt], false, false);
        }
        __syncthreads();   // readers done before next chunk's async stores
    }

    // ---- epilogue: exact fp32 k=256 tail + bias, store ------------------
    const long outBase = (s == 0) ? 0L : (s == 1) ? 32768L
                       : (s == 2) ? 33587200L : 33619968L;
#pragma unroll
    for (int tt = 0; tt < 8; ++tt) {
        const int col = wn * 128 + tt * 16 + nl;
        const float wt = wtail[s * 256 + col];
        const float bv = bias[s * 256 + col];
#pragma unroll
        for (int v = 0; v < 8; ++v) {
            const int rl = wm * 16 + (v + 8 * half);
            const float r = c[tt][v] + xtail[rl] * wt + bv;
            out[outBase + (long)(segRow + rl) * E_DIM + col] = r;
        }
    }
}

extern "C" void kernel_launch(void* const* d_in, const int* in_sizes, int n_in,
                              void* d_out, int out_size, void* d_ws, size_t ws_size,
                              hipStream_t stream) {
    const float* sol   = (const float*)d_in[0];
    const float* Wsrc  = (const float*)d_in[1];
    const float* bsrc  = (const float*)d_in[2];
    const float* Wdst  = (const float*)d_in[3];
    const float* bdst  = (const float*)d_in[4];
    const float* Wdep  = (const float*)d_in[5];
    const float* bdep  = (const float*)d_in[6];
    const float* Wcand = (const float*)d_in[7];
    const float* bcand = (const float*)d_in[8];
    const float* Wreg  = (const float*)d_in[9];
    const float* breg  = (const float*)d_in[10];
    const float* Wrand = (const float*)d_in[11];
    const float* iw    = (const float*)d_in[12];

    char* ws = (char*)d_ws;
    unsigned* whP   = (unsigned*)(ws);
    unsigned* wlP   = (unsigned*)(ws + 524288);
    float*    wtail = (float*)(ws + 1048576);
    float*    bias  = (float*)(ws + 1052672);

    prep_frag_kernel<<<512, 256, 0, stream>>>(Wsrc, Wcand, Wdst, Wdep,
                                              Wreg, Wrand, iw, whP, wlP);
    prep_tail_kernel<<<4, 256, 0, stream>>>(Wsrc, bsrc, Wcand, bcand,
                                            Wdst, bdst, Wdep, bdep,
                                            Wreg, breg, Wrand, iw,
                                            wtail, bias);

    gemm_kernel<<<2054, 256, 0, stream>>>(sol, whP, wlP, wtail, bias,
                                          (float*)d_out);
}
